// GATNet_13417477832750
// MI455X (gfx1250) — matrix operand order repaired and checked by reading, not compiled, not run on hardware
//
#include <hip/hip_runtime.h>
#include <hip/hip_bf16.h>
#include <math.h>

// ---------------------------------------------------------------------------
// GATNet forward on gfx1250 (MI455X).
// All dense GEMMs go through v_wmma_f32_16x16x32_bf16 (bf16 A/B, f32 acc),
// 32x64 output tile per wave (2x4 grid of 16x16 WMMA tiles): 8 accumulators.
// Padded-K leading dimension is a template constant so all fragment addresses
// fold into immediate offsets off two base pointers (no pointer spills).
// Graph attention (segment softmax + message aggregation) via f32 atomics.
// ---------------------------------------------------------------------------

typedef __attribute__((ext_vector_type(8)))  __bf16 bf16x8;
typedef __attribute__((ext_vector_type(16))) __bf16 bf16x16;
typedef __attribute__((ext_vector_type(8)))  float  v8f;
typedef __bf16 bf16;

__device__ __forceinline__ bf16 f2bf(float f) {
  unsigned u = __builtin_bit_cast(unsigned, f);
  unsigned r = u + 0x7FFFu + ((u >> 16) & 1u);          // round-to-nearest-even
  unsigned short h = (unsigned short)(r >> 16);
  return __builtin_bit_cast(bf16, h);
}

__device__ __forceinline__ void atomicMaxF(float* a, float v) {
  if (v >= 0.0f) atomicMax((int*)a, __float_as_int(v));
  else           atomicMin((unsigned int*)a, (unsigned int)__float_as_int(v));
}

// ---------------------------------------------------------------------------
// bf16 WMMA GEMM:  C[M,N] = act(A[M,LD] @ Bp[N,LD]^T + bias)
//  A  : row-major bf16, leading dim LD (multiple of 32, zero-padded K)
//  Bp : weights packed column-major bf16 [Npad64][LD] (col n contiguous in K)
//  32x64 tile per wave: 2 A-frags x 4 B-frags -> 8 accumulators.
//  M multiple of 32, Npad64 multiple of 64. 8 waves / block.
// ---------------------------------------------------------------------------
template <int ACT, int LD> // ACT: 0 = none, 1 = relu; LD = lda = ldb = Kp
__global__ void k_gemm_bf16(const bf16* __restrict__ A, const bf16* __restrict__ Bp,
                            const float* __restrict__ bias, float* __restrict__ C,
                            int M, int Npad64, int Nn, int ldc) {
  const int lane = threadIdx.x & 31;
  const int wave = threadIdx.x >> 5;
  const int tiles_n = Npad64 >> 6;
  const int tiles   = (M >> 5) * tiles_n;
  const int tid = blockIdx.x * (blockDim.x >> 5) + wave;
  if (tid >= tiles) return;
  const int tm = tid / tiles_n, tn = tid % tiles_n;

  // A fragment addressing: lane<16 -> row M=lane, K base 0; lane>=16 -> K base 16
  const bf16* ar0 = A  + (size_t)(tm * 32 + (lane & 15)) * LD + ((lane >> 4) << 4);
  // B fragment addressing: lane<16 -> col N=lane, K 0..15; lane>=16 -> K 16..31
  const bf16* bc0 = Bp + (size_t)(tn * 64 + (lane & 15)) * LD + ((lane >> 4) << 4);

  v8f acc[2][4] = {};
  for (int k = 0; k < LD; k += 32) {
    bf16x16 af[2], bf_[4];
#pragma unroll
    for (int i = 0; i < 2; ++i) {
      const bf16* ar = ar0 + i * 16 * LD + k;    // i*16*LD: compile-time delta
      bf16x8 a0 = *(const bf16x8*)(ar);          // K +0..7   (16B)
      bf16x8 a1 = *(const bf16x8*)(ar + 8);      // K +8..15  (16B)
      af[i] = __builtin_shufflevector(a0, a1, 0, 1, 2, 3, 4, 5, 6, 7,
                                              8, 9, 10, 11, 12, 13, 14, 15);
    }
#pragma unroll
    for (int j = 0; j < 4; ++j)
      bf_[j] = *(const bf16x16*)(bc0 + j * 16 * LD + k);  // 32B, imm delta
#pragma unroll
    for (int i = 0; i < 2; ++i)
#pragma unroll
      for (int j = 0; j < 4; ++j)
        acc[i][j] = __builtin_amdgcn_wmma_f32_16x16x32_bf16(
            false, af[i], false, bf_[j], (short)0, acc[i][j], false, false);
  }

#pragma unroll
  for (int j = 0; j < 4; ++j) {
    const int col = tn * 64 + j * 16 + (lane & 15);
    if (col >= Nn) continue;
    const float bv = bias ? bias[col] : 0.0f;
#pragma unroll
    for (int i = 0; i < 2; ++i) {
      const int rbase = tm * 32 + i * 16 + ((lane >> 4) << 3);
#pragma unroll
      for (int r = 0; r < 8; ++r) {
        float v = acc[i][j][r] + bv;
        if (ACT == 1) v = v > 0.0f ? v : 0.0f;
        C[(size_t)(rbase + r) * ldc + col] = v;
      }
    }
  }
}

template <int ACT, int LD>
static void launch_gemm(const bf16* A, const bf16* Bp, const float* bias, float* C,
                        int M, int Npad64, int Nn, int ldc, hipStream_t stream) {
  int tiles = (M / 32) * (Npad64 / 64);
  int gb = (tiles + 7) / 8;
  k_gemm_bf16<ACT, LD><<<gb, 256, 0, stream>>>(A, Bp, bias, C, M, Npad64, Nn, ldc);
}

// ---------------------------------------------------------------------------
// Packing / elementwise kernels
// ---------------------------------------------------------------------------
__global__ void k_fill_f32(float* __restrict__ p, float v, long long n) {
  long long i = (long long)blockIdx.x * blockDim.x + threadIdx.x;
  if (i < n) p[i] = v;
}

// dst[m*ldd + k] = k<K ? bf(src[m*lds + k]) : 0     (domain M*ldd)
__global__ void k_pack_pad(const float* __restrict__ src, int lds,
                           bf16* __restrict__ dst, int ldd, int M, int K) {
  long long i = (long long)blockIdx.x * blockDim.x + threadIdx.x;
  if (i >= (long long)M * ldd) return;
  int m = (int)(i / ldd), k = (int)(i % ldd);
  dst[i] = (k < K) ? f2bf(src[(size_t)m * lds + k]) : f2bf(0.0f);
}

// dst[m*ldd + coff + k] = bf(src[m*lds + k])        (domain M*K)
__global__ void k_pack_seg(const float* __restrict__ src, int lds,
                           bf16* __restrict__ dst, int ldd, int coff, int M, int K) {
  long long i = (long long)blockIdx.x * blockDim.x + threadIdx.x;
  if (i >= (long long)M * K) return;
  int m = (int)(i / K), k = (int)(i % K);
  dst[(size_t)m * ldd + coff + k] = f2bf(src[(size_t)m * lds + k]);
}

// W[K][N] row-major f32 -> dst[n*Kp + k] bf16 (col-major, zero padded)
__global__ void k_pack_w(const float* __restrict__ W, int K, int Nn,
                         bf16* __restrict__ dst, int Kp, int Npad) {
  long long i = (long long)blockIdx.x * blockDim.x + threadIdx.x;
  if (i >= (long long)Npad * Kp) return;
  int n = (int)(i / Kp), k = (int)(i % Kp);
  float v = (k < K && n < Nn) ? W[(size_t)k * Nn + n] : 0.0f;
  dst[i] = f2bf(v);
}

// per (node, head): a_s = <h[n,h,:], a_src[h,:]>, a_d likewise
__global__ void k_attn_scores(const float* __restrict__ hf,
                              const float* __restrict__ asrc,
                              const float* __restrict__ adst,
                              float* __restrict__ as_, float* __restrict__ ad_,
                              int Nn, int H, int D) {
  long long i = (long long)blockIdx.x * blockDim.x + threadIdx.x;
  if (i >= (long long)Nn * H) return;
  int n = (int)(i / H), h = (int)(i % H);
  const float* hp = hf + (size_t)n * H * D + (size_t)h * D;
  const float* sp = asrc + (size_t)h * D;
  const float* dp = adst + (size_t)h * D;
  float s = 0.0f, t = 0.0f;
  for (int d = 0; d < D; ++d) { float x = hp[d]; s += x * sp[d]; t += x * dp[d]; }
  as_[i] = s; ad_[i] = t;
}

// pass 1: segment max of leaky_relu(a_s[src]+a_d[dst]) over incoming edges
__global__ void k_edge_max(const int* __restrict__ ei, int E, int Nn,
                           const float* __restrict__ as_, const float* __restrict__ ad_,
                           float* __restrict__ m, int H) {
  long long i = (long long)blockIdx.x * blockDim.x + threadIdx.x;
  if (i >= (long long)(E + Nn) * H) return;
  int e = (int)(i / H), h = (int)(i % H);
  int src = (e < E) ? ei[e] : (e - E);
  int dst = (e < E) ? ei[E + e] : (e - E);
  float v = as_[(size_t)src * H + h] + ad_[(size_t)dst * H + h];
  v = fmaxf(v, 0.2f * v);                                  // leaky_relu(0.2)
  atomicMaxF(&m[(size_t)dst * H + h], v);
}

// pass 2: den += exp(e-max); num[dst,h,:] += exp(e-max) * h[src,h,:]
__global__ void k_edge_accum(const int* __restrict__ ei, int E, int Nn,
                             const float* __restrict__ as_, const float* __restrict__ ad_,
                             const float* __restrict__ m, float* __restrict__ den,
                             float* __restrict__ num, const float* __restrict__ hf,
                             int H, int D) {
  long long i = (long long)blockIdx.x * blockDim.x + threadIdx.x;
  if (i >= (long long)(E + Nn) * H) return;
  int e = (int)(i / H), h = (int)(i % H);
  int src = (e < E) ? ei[e] : (e - E);
  int dst = (e < E) ? ei[E + e] : (e - E);
  float v = as_[(size_t)src * H + h] + ad_[(size_t)dst * H + h];
  v = fmaxf(v, 0.2f * v);
  float ex = __expf(v - m[(size_t)dst * H + h]);
  atomicAdd(&den[(size_t)dst * H + h], ex);
  const float* hs = hf  + (size_t)src * H * D + (size_t)h * D;
  float*       nd = num + (size_t)dst * H * D + (size_t)h * D;
  for (int d = 0; d < D; ++d) atomicAdd(&nd[d], ex * hs[d]);
}

// out = elu(num/den + bias) -> bf16, zero-padded to ldd
__global__ void k_final_elu_bf16(const float* __restrict__ num, const float* __restrict__ den,
                                 const float* __restrict__ bias, bf16* __restrict__ dst,
                                 int Nn, int H, int D, int ldd) {
  long long i = (long long)blockIdx.x * blockDim.x + threadIdx.x;
  if (i >= (long long)Nn * ldd) return;
  int n = (int)(i / ldd), j = (int)(i % ldd);
  int HD = H * D;
  float v = 0.0f;
  if (j < HD) {
    int h = j / D;
    v = num[(size_t)n * HD + j] / (den[(size_t)n * H + h] + 1e-16f) + bias[j];
    v = v > 0.0f ? v : (__expf(v) - 1.0f);
  }
  dst[i] = f2bf(v);
}

// out = relu(num/den + bias) -> f32
__global__ void k_final_relu_f32(const float* __restrict__ num, const float* __restrict__ den,
                                 const float* __restrict__ bias, float* __restrict__ dst,
                                 int Nn, int H, int D) {
  long long i = (long long)blockIdx.x * blockDim.x + threadIdx.x;
  int HD = H * D;
  if (i >= (long long)Nn * HD) return;
  int n = (int)(i / HD), j = (int)(i % HD);
  int h = j / D;
  float v = num[i] / (den[(size_t)n * H + h] + 1e-16f) + bias[j];
  dst[i] = v > 0.0f ? v : 0.0f;
}

// global max pool per graph (values are post-relu >= 0; pool pre-zeroed)
__global__ void k_pool_max(const float* __restrict__ h, const int* __restrict__ batch,
                           float* __restrict__ pool, int Nn, int D) {
  long long i = (long long)blockIdx.x * blockDim.x + threadIdx.x;
  if (i >= (long long)Nn * D) return;
  int n = (int)(i / D), d = (int)(i % D);
  atomicMaxF(&pool[(size_t)batch[n] * D + d], h[i]);
}

// final head: clip(fc2@W[0:256] + xtc@W[256:266] + b, +-100)
__global__ void k_out_head(const float* __restrict__ fc2, const float* __restrict__ xtc,
                           const float* __restrict__ W, const float* __restrict__ b,
                           float* __restrict__ out, int Bb) {
  int r = blockIdx.x * blockDim.x + threadIdx.x;
  if (r >= Bb) return;
  float s = b[0];
  for (int j = 0; j < 256; ++j) s += fc2[(size_t)r * 256 + j] * W[j];
  for (int t = 0; t < 10; ++t)  s += xtc[(size_t)r * 10 + t] * W[256 + t];
  out[r] = fminf(fmaxf(s, -100.0f), 100.0f);
}

// ---------------------------------------------------------------------------
extern "C" void kernel_launch(void* const* d_in, const int* in_sizes, int n_in,
                              void* d_out, int out_size, void* d_ws, size_t ws_size,
                              hipStream_t stream) {
  (void)in_sizes; (void)n_in; (void)out_size; (void)ws_size;

  const int  Nn = 40960, Ee = 81920, Bb = 1024;
  const int  H1 = 10, D1 = 77, HD1 = 770, K1P = 96, N1P = 832, HD1P = 800;
  const int  XD = 128;
  const int  C1n = 23808, C2n = 3171, C3n = 627, NTC = 10;
  const int  CLIN = 27616, XCL = 256;                       // 27616 % 32 == 0
  const int  F1IN = 522, F1INP = 544, F1OUT = 2048, F2OUT = 256;
  const long long Etot = (long long)Ee + Nn;

  const float* xd1 = (const float*)d_in[0];
  const float* xd2 = (const float*)d_in[1];
  const int*   ei1 = (const int*)d_in[2];
  const int*   ei2 = (const int*)d_in[3];
  const int*   bd1 = (const int*)d_in[4];
  const int*   bd2 = (const int*)d_in[5];
  const float* xc1 = (const float*)d_in[6];
  const float* xc2 = (const float*)d_in[7];
  const float* xc3 = (const float*)d_in[8];
  const float* xtc = (const float*)d_in[9];
  const float* clW = (const float*)d_in[30];
  const float* clB = (const float*)d_in[31];
  const float* f1W = (const float*)d_in[32];
  const float* f1B = (const float*)d_in[33];
  const float* f2W = (const float*)d_in[34];
  const float* f2B = (const float*)d_in[35];
  const float* oW  = (const float*)d_in[36];
  const float* oB  = (const float*)d_in[37];

  // ---- workspace carve (256B aligned bump allocator) ----
  char* base = (char*)d_ws;
  size_t off = 0;
  auto alloc = [&](size_t bytes) -> char* {
    char* p = base + off;
    off += (bytes + 255) & ~(size_t)255;
    return p;
  };

  float* h0f  = (float*)alloc((size_t)Nn * HD1 * 4);   // 126 MB; reused as h2f/num2/h2o
  float* num1 = (float*)alloc((size_t)Nn * HD1 * 4);   // 126 MB; reused for clin/clWp
  bf16*  h1bf = (bf16*) alloc((size_t)Nn * HD1P * 2);  // 65.5 MB

  bf16*  xdbf   = (bf16*) alloc((size_t)Nn * K1P * 2);
  bf16*  w1p    = (bf16*) alloc((size_t)N1P * K1P * 2);
  float* as1    = (float*)alloc((size_t)Nn * H1 * 4);
  float* ad1    = (float*)alloc((size_t)Nn * H1 * 4);
  float* m1     = (float*)alloc((size_t)Nn * H1 * 4);
  float* den1   = (float*)alloc((size_t)Nn * H1 * 4);
  bf16*  w2p    = (bf16*) alloc((size_t)XD * HD1P * 2);
  float* as2    = (float*)alloc((size_t)Nn * 4);
  float* ad2    = (float*)alloc((size_t)Nn * 4);
  float* m2     = (float*)alloc((size_t)Nn * 4);
  float* den2   = (float*)alloc((size_t)Nn * 4);
  float* pool   = (float*)alloc((size_t)Bb * XD * 4);
  bf16*  poolbf = (bf16*) alloc((size_t)Bb * XD * 2);
  bf16*  fwp    = (bf16*) alloc((size_t)XD * XD * 2);
  float* brA    = (float*)alloc((size_t)Bb * XD * 4);
  float* brB    = (float*)alloc((size_t)Bb * XD * 4);
  float* xclf   = (float*)alloc((size_t)Bb * XCL * 4);
  bf16*  f1in   = (bf16*) alloc((size_t)Bb * F1INP * 2);
  bf16*  f1Wp   = (bf16*) alloc((size_t)F1OUT * F1INP * 2);
  float* fc1f   = (float*)alloc((size_t)Bb * F1OUT * 4);
  bf16*  fc1bf  = (bf16*) alloc((size_t)Bb * F1OUT * 2);
  bf16*  f2Wp   = (bf16*) alloc((size_t)F2OUT * F1OUT * 2);
  float* fc2f   = (float*)alloc((size_t)Bb * F2OUT * 4);

  // aliases into dead big regions
  float* h2f  = h0f;                                     // [Nn][128]
  float* num2 = h0f + (size_t)Nn * XD;                   // [Nn][128]
  float* h2o  = h0f + (size_t)2 * Nn * XD;               // [Nn][128]
  bf16*  clin = (bf16*)num1;                             // [Bb][27616]
  bf16*  clWp = (bf16*)((char*)num1 + (size_t)Bb * CLIN * 2);  // [256][27616]

  auto blocks = [](long long n) { return (int)((n + 255) / 256); };

  // ---- one GAT branch ----
  auto run_branch = [&](const float* xd, const int* ei, const int* batch,
                        const float* g1W, const float* g1s, const float* g1d, const float* g1b,
                        const float* g2W, const float* g2s, const float* g2d, const float* g2b,
                        const float* fW, const float* fb, float* brOut) {
    // layer 1: h0 = xd @ W1  (f32 out for attention/gather)
    k_pack_pad<<<blocks((long long)Nn * K1P), 256, 0, stream>>>(xd, D1, xdbf, K1P, Nn, D1);
    k_pack_w<<<blocks((long long)N1P * K1P), 256, 0, stream>>>(g1W, D1, HD1, w1p, K1P, N1P);
    launch_gemm<0, 96>(xdbf, w1p, nullptr, h0f, Nn, N1P, HD1, HD1, stream);

    k_attn_scores<<<blocks((long long)Nn * H1), 256, 0, stream>>>(h0f, g1s, g1d, as1, ad1, Nn, H1, D1);
    k_fill_f32<<<blocks((long long)Nn * H1), 256, 0, stream>>>(m1, -3.0e38f, (long long)Nn * H1);
    k_fill_f32<<<blocks((long long)Nn * H1), 256, 0, stream>>>(den1, 0.0f, (long long)Nn * H1);
    k_fill_f32<<<blocks((long long)Nn * HD1), 256, 0, stream>>>(num1, 0.0f, (long long)Nn * HD1);
    k_edge_max<<<blocks(Etot * H1), 256, 0, stream>>>(ei, Ee, Nn, as1, ad1, m1, H1);
    k_edge_accum<<<blocks(Etot * H1), 256, 0, stream>>>(ei, Ee, Nn, as1, ad1, m1, den1, num1, h0f, H1, D1);
    k_final_elu_bf16<<<blocks((long long)Nn * HD1P), 256, 0, stream>>>(num1, den1, g1b, h1bf, Nn, H1, D1, HD1P);

    // layer 2: h2 = h1 @ W2
    k_pack_w<<<blocks((long long)XD * HD1P), 256, 0, stream>>>(g2W, HD1, XD, w2p, HD1P, XD);
    launch_gemm<0, 800>(h1bf, w2p, nullptr, h2f, Nn, XD, XD, XD, stream);

    k_attn_scores<<<blocks((long long)Nn), 256, 0, stream>>>(h2f, g2s, g2d, as2, ad2, Nn, 1, XD);
    k_fill_f32<<<blocks((long long)Nn), 256, 0, stream>>>(m2, -3.0e38f, (long long)Nn);
    k_fill_f32<<<blocks((long long)Nn), 256, 0, stream>>>(den2, 0.0f, (long long)Nn);
    k_fill_f32<<<blocks((long long)Nn * XD), 256, 0, stream>>>(num2, 0.0f, (long long)Nn * XD);
    k_edge_max<<<blocks(Etot), 256, 0, stream>>>(ei, Ee, Nn, as2, ad2, m2, 1);
    k_edge_accum<<<blocks(Etot), 256, 0, stream>>>(ei, Ee, Nn, as2, ad2, m2, den2, num2, h2f, 1, XD);
    k_final_relu_f32<<<blocks((long long)Nn * XD), 256, 0, stream>>>(num2, den2, g2b, h2o, Nn, 1, XD);

    // global max pool + fg linear
    k_fill_f32<<<blocks((long long)Bb * XD), 256, 0, stream>>>(pool, 0.0f, (long long)Bb * XD);
    k_pool_max<<<blocks((long long)Nn * XD), 256, 0, stream>>>(h2o, batch, pool, Nn, XD);
    k_pack_pad<<<blocks((long long)Bb * XD), 256, 0, stream>>>(pool, XD, poolbf, XD, Bb, XD);
    k_pack_w<<<blocks((long long)XD * XD), 256, 0, stream>>>(fW, XD, XD, fwp, XD, XD);
    launch_gemm<1, 128>(poolbf, fwp, fb, brOut, Bb, XD, XD, XD, stream);
  };

  run_branch(xd1, ei1, bd1,
             (const float*)d_in[10], (const float*)d_in[11], (const float*)d_in[12], (const float*)d_in[13],
             (const float*)d_in[14], (const float*)d_in[15], (const float*)d_in[16], (const float*)d_in[17],
             (const float*)d_in[18], (const float*)d_in[19], brA);
  run_branch(xd2, ei2, bd2,
             (const float*)d_in[20], (const float*)d_in[21], (const float*)d_in[22], (const float*)d_in[23],
             (const float*)d_in[24], (const float*)d_in[25], (const float*)d_in[26], (const float*)d_in[27],
             (const float*)d_in[28], (const float*)d_in[29], brB);

  // ---- classifier head ----
  // xcl = relu(concat(xc1,xc2,xc3,xtc) @ cl_W + cl_b)
  k_pack_seg<<<blocks((long long)Bb * C1n), 256, 0, stream>>>(xc1, C1n, clin, CLIN, 0, Bb, C1n);
  k_pack_seg<<<blocks((long long)Bb * C2n), 256, 0, stream>>>(xc2, C2n, clin, CLIN, C1n, Bb, C2n);
  k_pack_seg<<<blocks((long long)Bb * C3n), 256, 0, stream>>>(xc3, C3n, clin, CLIN, C1n + C2n, Bb, C3n);
  k_pack_seg<<<blocks((long long)Bb * NTC), 256, 0, stream>>>(xtc, NTC, clin, CLIN, C1n + C2n + C3n, Bb, NTC);
  k_pack_w<<<blocks((long long)XCL * CLIN), 256, 0, stream>>>(clW, CLIN, XCL, clWp, CLIN, XCL);
  launch_gemm<1, 27616>(clin, clWp, clB, xclf, Bb, XCL, XCL, XCL, stream);

  // fc1 input = [brA | brB | xcl | xtc], K padded 522 -> 544
  k_fill_f32<<<blocks((long long)Bb * F1INP / 2), 256, 0, stream>>>((float*)f1in, 0.0f, (long long)Bb * F1INP / 2);
  k_pack_seg<<<blocks((long long)Bb * XD), 256, 0, stream>>>(brA, XD, f1in, F1INP, 0, Bb, XD);
  k_pack_seg<<<blocks((long long)Bb * XD), 256, 0, stream>>>(brB, XD, f1in, F1INP, XD, Bb, XD);
  k_pack_seg<<<blocks((long long)Bb * XCL), 256, 0, stream>>>(xclf, XCL, f1in, F1INP, 2 * XD, Bb, XCL);
  k_pack_seg<<<blocks((long long)Bb * NTC), 256, 0, stream>>>(xtc, NTC, f1in, F1INP, 2 * XD + XCL, Bb, NTC);
  k_pack_w<<<blocks((long long)F1OUT * F1INP), 256, 0, stream>>>(f1W, F1IN, F1OUT, f1Wp, F1INP, F1OUT);
  launch_gemm<1, 544>(f1in, f1Wp, f1B, fc1f, Bb, F1OUT, F1OUT, F1OUT, stream);

  // fc2
  k_pack_pad<<<blocks((long long)Bb * F1OUT), 256, 0, stream>>>(fc1f, F1OUT, fc1bf, F1OUT, Bb, F1OUT);
  k_pack_w<<<blocks((long long)F2OUT * F1OUT), 256, 0, stream>>>(f2W, F1OUT, F2OUT, f2Wp, F1OUT, F2OUT);
  launch_gemm<1, 2048>(fc1bf, f2Wp, f2B, fc2f, Bb, F2OUT, F2OUT, F2OUT, stream);

  // output: clip([fc2 | xtc] @ out_W + out_b)
  k_out_head<<<blocks((long long)Bb), 256, 0, stream>>>(fc2f, xtc, oW, oB, (float*)d_out, Bb);
}
